// LinformerAttention_77146202571058
// MI455X (gfx1250) — matrix operand order repaired and checked by reading, not compiled, not run on hardware
//
#include <hip/hip_runtime.h>

// ---------- types ----------
typedef __attribute__((ext_vector_type(16))) __bf16 bf16x16;
typedef __attribute__((ext_vector_type(8)))  float  f32x8;
typedef __attribute__((ext_vector_type(4)))  float  f32x4;
typedef __attribute__((ext_vector_type(4)))  unsigned int u32x4;
typedef __attribute__((ext_vector_type(4)))  int i32x4;

union Frag { u32x4 q[2]; bf16x16 v; };
union U32B { unsigned int u; __bf16 b[2]; };
union B4   { unsigned long long u; __bf16 b[4]; };

#define T_SEQ 2048
#define C_DIM 1024
#define C3    3072
#define HEADS 16
#define DHEAD 64
#define PROJ  256

// ---------- CDNA5 async global->LDS path (guarded; falls back to manual copy) ----------
#if defined(__AMDGCN__) && __has_builtin(__builtin_amdgcn_global_load_async_to_lds_b128)
#define HAS_ASYNC 1
typedef __attribute__((address_space(1))) i32x4* gi32x4p;  // pointer to AS1 int4
typedef __attribute__((address_space(3))) i32x4* li32x4p;  // pointer to AS3 int4
#define ASYNC_CP_B128(gsrc, ldst)                                              \
  __builtin_amdgcn_global_load_async_to_lds_b128(                              \
      (gi32x4p)(gsrc), (li32x4p)(ldst), 0, 0)
#define ASYNC_WAIT() asm volatile("s_wait_asynccnt 0x0" ::: "memory")
#else
#define HAS_ASYNC 0
#define ASYNC_WAIT()
#endif

#define WMMA_BF16(a, b, c)                                                     \
  __builtin_amdgcn_wmma_f32_16x16x32_bf16(false, (a), false, (b), (short)0,    \
                                          (c), false, false)

// ---------- f32 -> bf16 cast (b128 in / b64 out) ----------
__global__ void cast_f32_bf16(const float* __restrict__ in, __bf16* __restrict__ out, long long n) {
  long long i = (long long)blockIdx.x * blockDim.x + threadIdx.x;
  long long stride = (long long)gridDim.x * blockDim.x;
  long long n4 = n >> 2;
  const f32x4* in4 = (const f32x4*)in;
  unsigned long long* out4 = (unsigned long long*)out;
  for (; i < n4; i += stride) {
    f32x4 v = in4[i];
    B4 p;
    p.b[0] = (__bf16)v.x; p.b[1] = (__bf16)v.y;
    p.b[2] = (__bf16)v.z; p.b[3] = (__bf16)v.w;
    out4[i] = p.u;
  }
}

// ---------- generic batched WMMA GEMM ----------
// C[z] = A[z] (MxK row-major bf16) @ B[z] (KxN row-major bf16); b=z/H, h=z%H strides.
// Block tile 128(M) x 128(N) x 32(K), double-buffered LDS, async A staging.
// 256 threads = 8 waves; wave w owns rows w*16..w*16+15 and all 8 N-tiles.
__global__ __launch_bounds__(256) void gemm_bf16(
    const __bf16* __restrict__ A, long long lda, long long sAb, long long sAh,
    const __bf16* __restrict__ Bm, long long ldb, long long sBb, long long sBh,
    void* __restrict__ Cv, long long ldc, long long sC,
    int M, int N, int K, int H, int out_f32)
{
  __shared__ __align__(16) __bf16 As[2][128][40];  // [m][k], rows 80B (16B multiple)
  __shared__ __align__(16) __bf16 Bs[2][128][40];  // [n][k] transposed stage
  const int tid  = threadIdx.x;
  const int w    = tid >> 5;
  const int lane = tid & 31;
  const int ln   = lane & 15;
  const int hh   = lane >> 4;
  const int z  = blockIdx.z;
  const int bb = z / H, h = z % H;
  A  += (size_t)bb * sAb + (size_t)h * sAh;
  Bm += (size_t)bb * sBb + (size_t)h * sBh;
  const long long row0 = (long long)blockIdx.y * 128;
  const long long col0 = (long long)blockIdx.x * 128;
  int ntAct = (int)((N - col0 + 15) >> 4);
  if (ntAct > 8) ntAct = 8;

  auto stageA = [&](int kk, int buf) {
#if HAS_ASYNC
#pragma unroll
    for (int i = 0; i < 2; ++i) {           // 128 rows x 4 chunks of 16B
      int idx = tid + i * 256;
      int r = idx >> 2, cp = idx & 3;
      ASYNC_CP_B128(A + (row0 + r) * lda + kk + cp * 8, &As[buf][r][cp * 8]);
    }
#else
#pragma unroll
    for (int i = 0; i < 8; ++i) {
      int idx = tid + i * 256;
      int r = idx >> 4, cp = idx & 15;
      *(unsigned int*)&As[buf][r][cp * 2] =
          *(const unsigned int*)(A + (row0 + r) * lda + kk + cp * 2);
    }
#endif
  };
  auto stageB = [&](int kk, int buf) {      // 32 x 128, transpose to [n][k]
#pragma unroll
    for (int i = 0; i < 8; ++i) {
      int idx = tid + i * 256;
      int kr = idx >> 6, np = idx & 63;
      U32B u;
      u.u = *(const unsigned int*)(Bm + (long long)(kk + kr) * ldb + col0 + np * 2);
      Bs[buf][np * 2][kr]     = u.b[0];
      Bs[buf][np * 2 + 1][kr] = u.b[1];
    }
  };
  auto loadB = [&](Frag& f, int cur, int tn) {
    f.q[0] = *(const u32x4*)&Bs[cur][tn * 16 + ln][hh * 8];
    f.q[1] = *(const u32x4*)&Bs[cur][tn * 16 + ln][hh * 8 + 16];
  };

  const f32x8 vzero = {0.f,0.f,0.f,0.f,0.f,0.f,0.f,0.f};
  f32x8 acc[8];
#pragma unroll
  for (int t = 0; t < 8; ++t) acc[t] = vzero;

  stageA(0, 0);
  stageB(0, 0);
  ASYNC_WAIT();
  __syncthreads();

  if (ntAct == 8) {
    // ---- hot path: branch-free inner loop, pipelined B-fragment loads ----
    for (int kk = 0; kk < K; kk += 32) {
      const int cur = (kk >> 5) & 1;
      if (kk + 32 < K) {
        stageA(kk + 32, cur ^ 1);
        stageB(kk + 32, cur ^ 1);
      }
      Frag fa;
      fa.q[0] = *(const u32x4*)&As[cur][w * 16 + ln][hh * 8];
      fa.q[1] = *(const u32x4*)&As[cur][w * 16 + ln][hh * 8 + 16];
      Frag fb[2];
      loadB(fb[0], cur, 0);
#pragma unroll
      for (int tn = 0; tn < 8; ++tn) {
        if (tn < 7) loadB(fb[(tn + 1) & 1], cur, tn + 1);  // prefetch next frag
        acc[tn] = WMMA_BF16(fa.v, fb[tn & 1].v, acc[tn]);
      }
      ASYNC_WAIT();
      __syncthreads();
    }
  } else {
    // ---- partial-N path (proj GEMM, N=64) ----
    for (int kk = 0; kk < K; kk += 32) {
      const int cur = (kk >> 5) & 1;
      if (kk + 32 < K) {
        stageA(kk + 32, cur ^ 1);
        stageB(kk + 32, cur ^ 1);
      }
      Frag fa;
      fa.q[0] = *(const u32x4*)&As[cur][w * 16 + ln][hh * 8];
      fa.q[1] = *(const u32x4*)&As[cur][w * 16 + ln][hh * 8 + 16];
#pragma unroll
      for (int tn = 0; tn < 8; ++tn) {
        if (tn < ntAct) {
          Frag fb;
          loadB(fb, cur, tn);
          acc[tn] = WMMA_BF16(fa.v, fb.v, acc[tn]);
        }
      }
      ASYNC_WAIT();
      __syncthreads();
    }
  }

#pragma unroll
  for (int tn = 0; tn < 8; ++tn) {
    if (tn < ntAct) {
#pragma unroll
      for (int r = 0; r < 8; ++r) {
        long long row = row0 + w * 16 + r + 8 * hh;
        long long col = col0 + tn * 16 + ln;
        float v = acc[tn][r];
        if (out_f32) ((float*)Cv)[(size_t)z * sC + row * ldc + col] = v;
        else         ((__bf16*)Cv)[(size_t)z * sC + row * ldc + col] = (__bf16)v;
      }
    }
  }
}

// ---------- fused attention: S = q@kpT * 1/8, tril mask, softmax, y = P@vp ----------
__global__ __launch_bounds__(256) void attn_fused(
    const __bf16* __restrict__ qkvb, const __bf16* __restrict__ kpb,
    const __bf16* __restrict__ vpb, __bf16* __restrict__ yb)
{
  __shared__ __align__(16) union {
    __bf16 kps[256][72];                                      // phase A: [n=p][k=d]
    struct { __bf16 pms[128][136]; __bf16 vps[64][136]; } p2; // phase B (K halves)
  } sm;

  const int tid  = threadIdx.x;
  const int w    = tid >> 5;
  const int lane = tid & 31;
  const int ln   = lane & 15;
  const int hh   = lane >> 4;
  const int bh = blockIdx.y;
  const int b  = bh / HEADS, h = bh % HEADS;
  const int t0 = blockIdx.x * 128;
  const f32x8 vzero = {0.f,0.f,0.f,0.f,0.f,0.f,0.f,0.f};

  // k_proj tile: contiguous 256x64 bf16 -> LDS (async DMA when available)
  {
    const __bf16* src = kpb + (size_t)bh * (PROJ * DHEAD);
#if HAS_ASYNC
#pragma unroll
    for (int i = 0; i < 8; ++i) {          // 2048 chunks of 16B
      int idx = tid + i * 256;
      int r = idx >> 3, cp = idx & 7;
      ASYNC_CP_B128(src + (size_t)idx * 8, &sm.kps[r][cp * 8]);
    }
    ASYNC_WAIT();
#else
    const unsigned int* s32 = (const unsigned int*)src;
#pragma unroll
    for (int i = 0; i < 32; ++i) {
      int idx = tid + i * 256;
      int r = idx >> 5, cp = idx & 31;
      *(unsigned int*)&sm.kps[r][cp * 2] = s32[idx];
    }
#endif
  }
  // q A-fragments straight from global
  Frag a0, a1;
  {
    const __bf16* qp = qkvb + (size_t)(b * T_SEQ + t0 + w * 16 + ln) * C3 + (size_t)h * DHEAD;
    a0.q[0] = *(const u32x4*)(qp + hh * 8);
    a0.q[1] = *(const u32x4*)(qp + hh * 8 + 16);
    a1.q[0] = *(const u32x4*)(qp + 32 + hh * 8);
    a1.q[1] = *(const u32x4*)(qp + 32 + hh * 8 + 16);
  }
  __syncthreads();

  // Phase A: S tile 128x256 in accumulators (16 N-tiles/wave, K=64 -> 2 wmma each)
  auto loadK = [&](Frag& f0, Frag& f1, int tn) {
    f0.q[0] = *(const u32x4*)&sm.kps[tn * 16 + ln][hh * 8];
    f0.q[1] = *(const u32x4*)&sm.kps[tn * 16 + ln][hh * 8 + 16];
    f1.q[0] = *(const u32x4*)&sm.kps[tn * 16 + ln][32 + hh * 8];
    f1.q[1] = *(const u32x4*)&sm.kps[tn * 16 + ln][32 + hh * 8 + 16];
  };
  f32x8 sacc[16];
#pragma unroll
  for (int tn = 0; tn < 16; ++tn) sacc[tn] = vzero;
  {
    Frag c0, c1, n0, n1;
    loadK(c0, c1, 0);
#pragma unroll
    for (int tn = 0; tn < 16; ++tn) {
      if (tn < 15) loadK(n0, n1, tn + 1);   // prefetch next tile's fragments
      sacc[tn] = WMMA_BF16(a0.v, c0.v, sacc[tn]);
      sacc[tn] = WMMA_BF16(a1.v, c1.v, sacc[tn]);
      c0 = n0; c1 = n1;
    }
  }

  // scale + tril mask + row softmax (rows live in one 16-lane half)
  const float scale = 0.125f; // 1/sqrt(64)
#pragma unroll
  for (int r = 0; r < 8; ++r) {
    int trow = t0 + w * 16 + r + 8 * hh;
    float mx = -1e30f;
#pragma unroll
    for (int tn = 0; tn < 16; ++tn) {
      int pcol = tn * 16 + ln;
      float v = sacc[tn][r] * scale;
      if (pcol > trow) v = -1e30f;
      sacc[tn][r] = v;
      mx = fmaxf(mx, v);
    }
#pragma unroll
    for (int off = 1; off < 16; off <<= 1) mx = fmaxf(mx, __shfl_xor(mx, off, 32));
    float sum = 0.f;
#pragma unroll
    for (int tn = 0; tn < 16; ++tn) {
      float p = __expf(sacc[tn][r] - mx);
      sacc[tn][r] = p;
      sum += p;
    }
#pragma unroll
    for (int off = 1; off < 16; off <<= 1) sum += __shfl_xor(sum, off, 32);
    float inv = 1.0f / sum;
#pragma unroll
    for (int tn = 0; tn < 16; ++tn) sacc[tn][r] *= inv;
  }
  __syncthreads();  // all waves done with kps before union repurpose

  // Phase B: y = P(128x256) @ v_proj(256x64), K split in two 128 halves
  auto loadV = [&](Frag& f, int ks, int tn) {
    f.q[0] = *(const u32x4*)&sm.p2.vps[tn * 16 + ln][ks * 32 + hh * 8];
    f.q[1] = *(const u32x4*)&sm.p2.vps[tn * 16 + ln][ks * 32 + hh * 8 + 16];
  };
  f32x8 acc2[4];
#pragma unroll
  for (int t = 0; t < 4; ++t) acc2[t] = vzero;

#pragma unroll
  for (int half = 0; half < 2; ++half) {
#pragma unroll
    for (int tl = 0; tl < 8; ++tl) {       // probs -> LDS A-layout [m][k]
      int tn = half * 8 + tl;
#pragma unroll
      for (int r = 0; r < 8; ++r)
        sm.p2.pms[w * 16 + r + 8 * hh][tl * 16 + ln] = (__bf16)sacc[tn][r];
    }
    {
      const unsigned int* vsrc = (const unsigned int*)(
          vpb + (size_t)bh * (PROJ * DHEAD) + (size_t)half * 128 * DHEAD);
#pragma unroll
      for (int i = 0; i < 16; ++i) {       // v_proj half, transposed to [n=d][k=p]
        int idx = tid + i * 256;
        int kl = idx >> 5, dp = idx & 31;
        U32B u; u.u = vsrc[idx];
        sm.p2.vps[dp * 2][kl]     = u.b[0];
        sm.p2.vps[dp * 2 + 1][kl] = u.b[1];
      }
    }
    __syncthreads();
#pragma unroll
    for (int ks = 0; ks < 4; ++ks) {
      Frag fa;
      fa.q[0] = *(const u32x4*)&sm.p2.pms[w * 16 + ln][ks * 32 + hh * 8];
      fa.q[1] = *(const u32x4*)&sm.p2.pms[w * 16 + ln][ks * 32 + hh * 8 + 16];
      Frag fb[2];
      loadV(fb[0], ks, 0);
#pragma unroll
      for (int tn = 0; tn < 4; ++tn) {
        if (tn < 3) loadV(fb[(tn + 1) & 1], ks, tn + 1);
        acc2[tn] = WMMA_BF16(fa.v, fb[tn & 1].v, acc2[tn]);
      }
    }
    __syncthreads();
  }

  // y -> [B,T,C] bf16 (head-interleaved) for final projection GEMM
#pragma unroll
  for (int tn = 0; tn < 4; ++tn) {
#pragma unroll
    for (int r = 0; r < 8; ++r) {
      int t = t0 + w * 16 + r + 8 * hh;
      int d = tn * 16 + ln;
      yb[(size_t)(b * T_SEQ + t) * C_DIM + (size_t)h * DHEAD + d] = (__bf16)acc2[tn][r];
    }
  }
}

// ---------- host ----------
extern "C" void kernel_launch(void* const* d_in, const int* in_sizes, int n_in,
                              void* d_out, int out_size, void* d_ws, size_t ws_size,
                              hipStream_t stream) {
  const float* x  = (const float*)d_in[0];
  const float* Wa = (const float*)d_in[1];
  const float* Wp = (const float*)d_in[2];
  const float* E  = (const float*)d_in[3];
  const float* Fp = (const float*)d_in[4];
  float* out = (float*)d_out;

  // workspace layout (bf16 elements); total ~208 MB
  __bf16* ws   = (__bf16*)d_ws;
  __bf16* xb   = ws;                      // 16384x1024
  __bf16* Wab  = ws + 16777216LL;         // 1024x3072
  __bf16* Wpb  = ws + 19922944LL;         // 1024x1024
  __bf16* Eb   = ws + 20971520LL;         // 16x256x2048
  __bf16* Fb   = ws + 29360128LL;         // 16x256x2048
  __bf16* qkvb = ws + 37748736LL;         // 16384x3072
  __bf16* kpb  = ws + 88080384LL;         // 128x256x64
  __bf16* vpb  = ws + 90177536LL;         // 128x256x64
  __bf16* yb   = ws + 92274688LL;         // 16384x1024

  cast_f32_bf16<<<1024, 256, 0, stream>>>(x,  xb,  16777216LL);
  cast_f32_bf16<<<512,  256, 0, stream>>>(Wa, Wab, 3145728LL);
  cast_f32_bf16<<<256,  256, 0, stream>>>(Wp, Wpb, 1048576LL);
  cast_f32_bf16<<<1024, 256, 0, stream>>>(E,  Eb,  8388608LL);
  cast_f32_bf16<<<1024, 256, 0, stream>>>(Fp, Fb,  8388608LL);

  // 1) qkv = x @ W_attn   (M=16384, N=3072, K=1024)
  {
    dim3 g(24, 128, 1);
    gemm_bf16<<<g, 256, 0, stream>>>(xb, 1024, 0, 0,
                                     Wab, 3072, 0, 0,
                                     qkvb, 3072, 0,
                                     16384, 3072, 1024, 1, 0);
  }
  // 2) k_proj = E[h] @ k[b,h], v_proj = F[h] @ v[b,h]  (M=256, N=64, K=2048, z=b*H+h)
  {
    dim3 g(1, 2, 128);
    gemm_bf16<<<g, 256, 0, stream>>>(Eb, 2048, 0, 524288LL,
                                     qkvb + 1024, 3072, 6291456LL, 64,
                                     kpb, 64, 16384LL,
                                     256, 64, 2048, 16, 0);
    gemm_bf16<<<g, 256, 0, stream>>>(Fb, 2048, 0, 524288LL,
                                     qkvb + 2048, 3072, 6291456LL, 64,
                                     vpb, 64, 16384LL,
                                     256, 64, 2048, 16, 0);
  }
  // 3+4) fused score-GEMM / mask / softmax / prob-GEMM
  {
    dim3 g(16, 128, 1);
    attn_fused<<<g, 256, 0, stream>>>(qkvb, kpb, vpb, yb);
  }
  // 5) out = y @ W_proj  (f32 output)
  {
    dim3 g(8, 128, 1);
    gemm_bf16<<<g, 256, 0, stream>>>(yb, 1024, 0, 0,
                                     Wpb, 1024, 0, 0,
                                     out, 1024, 0,
                                     16384, 1024, 1024, 1, 1);
  }
}
